// SonicMoE_84868553769175
// MI455X (gfx1250) — matrix operand order repaired and checked by reading, not compiled, not run on hardware
//
#include <hip/hip_runtime.h>
#include <math.h>

typedef __attribute__((ext_vector_type(16))) __bf16 v16bf;
typedef __attribute__((ext_vector_type(8)))  float  v8f;
typedef __attribute__((ext_vector_type(4)))  unsigned int v4u;

union FragBF { v4u u[2]; v16bf v; };

constexpr int T = 1024, C = 1024, E = 8, I = 1024;
constexpr int BM = 128, BN = 128, BK = 32;
constexpr int LDK = 40;          // padded LDS row stride (bf16 elems): 80B rows -> conflict-free b128 lanes
constexpr int ATILE = BM * LDK;  // elements per A buffer
constexpr int BTILE = BN * LDK;  // elements per B buffer

// fp32 -> bf16, round-half-up (cheap: +0x8000, take high half)
__device__ __forceinline__ unsigned short f2bf(float f) {
  return (unsigned short)((__float_as_uint(f) + 0x8000u) >> 16);
}

#if defined(__has_builtin) && __has_builtin(__builtin_amdgcn_perm)
#define MOE_HAVE_PERM 1
#endif

// pack two fp32 into packed bf16 pair: one v_perm_b32 grabs both high halves
__device__ __forceinline__ unsigned int pk2(float a, float b) {
  unsigned ua = __float_as_uint(a) + 0x8000u;
  unsigned ub = __float_as_uint(b) + 0x8000u;
#ifdef MOE_HAVE_PERM
  return __builtin_amdgcn_perm(ub, ua, 0x07060302u);  // {ub[3],ub[2],ua[3],ua[2]}
#else
  return (ua >> 16) | (ub & 0xffff0000u);
#endif
}

__device__ __forceinline__ v8f wmma_bf16(v16bf a, v16bf b, v8f c) {
  return __builtin_amdgcn_wmma_f32_16x16x32_bf16(false, a, false, b, (short)0, c, false, false);
}

// CDNA5 async copy: global -> LDS, tracked on ASYNCcnt (bypasses VGPRs).
__device__ __forceinline__ void async_ld_b128(unsigned lds_byte_off, const void* gptr) {
  asm volatile("global_load_async_to_lds_b128 %0, %1, off"
               :: "v"(lds_byte_off), "v"((unsigned long long)(size_t)gptr)
               : "memory");
}
__device__ __forceinline__ void wait_async0() {
  asm volatile("s_wait_asynccnt 0x0" ::: "memory");
}
__device__ __forceinline__ unsigned lds_off(const void* p) {
  // flat shared-aperture address keeps the LDS byte offset in the low 32 bits
  return (unsigned)(size_t)p;
}

// ---------------- fp32 -> bf16 conversion of x ----------------
__global__ __launch_bounds__(256) void moe_cvt_kernel(const float* __restrict__ x,
                                                      unsigned short* __restrict__ xb) {
  int i = blockIdx.x * 256 + threadIdx.x;
  if (i < T * C / 4) {
    float4 v = ((const float4*)x)[i];
    uint2 p; p.x = pk2(v.x, v.y); p.y = pk2(v.z, v.w);
    ((uint2*)xb)[i] = p;
  }
}

// ---------------- router: logits -> softmax -> top-2, build expert lists ----------------
__global__ __launch_bounds__(256) void moe_router_kernel(const float* __restrict__ x,
                                                         const float* __restrict__ rw,
                                                         int* __restrict__ counts,
                                                         int* __restrict__ idxl,
                                                         float* __restrict__ gvl) {
  __shared__ float sRW[E * C];  // 32 KB
  const int tid = threadIdx.x;
  for (int i = tid; i < E * C / 4; i += 256)
    ((float4*)sRW)[i] = ((const float4*)rw)[i];
  __syncthreads();

  const int lane = tid & 31;
  const int wv   = tid >> 5;           // one wave32 per token
  const int t    = blockIdx.x * 8 + wv;

  float acc[E];
#pragma unroll
  for (int e = 0; e < E; ++e) acc[e] = 0.0f;

  const float* xr = x + (size_t)t * C;
  for (int i = 0; i < C / 32; ++i) {
    int c = lane + i * 32;
    float xv = xr[c];
#pragma unroll
    for (int e = 0; e < E; ++e) acc[e] += xv * sRW[e * C + c];
  }
#pragma unroll
  for (int e = 0; e < E; ++e)
    for (int off = 16; off > 0; off >>= 1)
      acc[e] += __shfl_xor(acc[e], off, 32);

  if (lane == 0) {
    float mx = acc[0];
#pragma unroll
    for (int e = 1; e < E; ++e) mx = fmaxf(mx, acc[e]);
    float p[E], s = 0.0f;
#pragma unroll
    for (int e = 0; e < E; ++e) { p[e] = __expf(acc[e] - mx); s += p[e]; }
    float inv = 1.0f / s;
    int i1 = 0;
#pragma unroll
    for (int e = 1; e < E; ++e) if (p[e] > p[i1]) i1 = e;
    int i2 = (i1 == 0) ? 1 : 0;
#pragma unroll
    for (int e = 0; e < E; ++e) if (e != i1 && p[e] > p[i2]) i2 = e;
    int pos1 = atomicAdd(&counts[i1], 1);
    idxl[i1 * T + pos1] = t; gvl[i1 * T + pos1] = p[i1] * inv;
    int pos2 = atomicAdd(&counts[i2], 1);
    idxl[i2 * T + pos2] = t; gvl[i2 * T + pos2] = p[i2] * inv;
  }
}

// ---------------- GEMM1: h = gather(x) @ w_in^T, SwiGLU -> a_buf (bf16) ----------------
__global__ __launch_bounds__(256) void moe_gemm1_kernel(const unsigned short* __restrict__ xb,
                                                        const float* __restrict__ w_in,
                                                        const float* __restrict__ b_in,
                                                        const int* __restrict__ counts,
                                                        const int* __restrict__ idxl,
                                                        unsigned short* __restrict__ a_buf) {
  const int e  = blockIdx.z;
  const int m0 = blockIdx.x * BM;
  const int n0 = blockIdx.y * BN;
  const int Ne = counts[e];
  if (m0 >= Ne) return;  // uniform data-dependent early exit

  __shared__ __align__(16) unsigned short sA[2 * ATILE];
  __shared__ __align__(16) unsigned short sG[2 * BTILE];
  __shared__ __align__(16) unsigned short sU[2 * BTILE];

  const int tid  = threadIdx.x;
  const int lane = tid & 31;
  const int wv   = tid >> 5;
  const int wm   = (wv >> 1) << 5;   // 0,32,64,96
  const int wn   = (wv & 1) << 6;    // 0,64
  const int ar   = tid >> 1;         // staging row 0..127
  const int ac   = (tid & 1) << 4;   // staging col 0/16

  const bool av = (m0 + ar) < Ne;
  const int tok = av ? idxl[e * T + m0 + ar] : 0;   // pad rows re-read a valid row; discarded later
  const unsigned short* aSrc = xb + (size_t)tok * C;
  const float* gRow = w_in + ((size_t)e * 2 * I + (n0 + ar)) * C;
  const float* uRow = w_in + ((size_t)e * 2 * I + I + n0 + ar) * C;

  const unsigned aDst0 = lds_off(&sA[ar * LDK + ac]);

  struct Pre { v4u g0, g1, u0, u1; };
  auto issueA = [&](int buf, int k0) {
    unsigned d = aDst0 + (unsigned)(buf * ATILE * 2);
    const unsigned short* g = aSrc + k0 + ac;
    async_ld_b128(d,      g);
    async_ld_b128(d + 16, g + 8);
  };
  auto loadGU = [&](int k0) {
    Pre r;
    const float4* gs = (const float4*)(gRow + k0 + ac);
    float4 f0 = gs[0], f1 = gs[1], f2 = gs[2], f3 = gs[3];
    r.g0 = { pk2(f0.x, f0.y), pk2(f0.z, f0.w), pk2(f1.x, f1.y), pk2(f1.z, f1.w) };
    r.g1 = { pk2(f2.x, f2.y), pk2(f2.z, f2.w), pk2(f3.x, f3.y), pk2(f3.z, f3.w) };
    const float4* us = (const float4*)(uRow + k0 + ac);
    float4 h0 = us[0], h1 = us[1], h2 = us[2], h3 = us[3];
    r.u0 = { pk2(h0.x, h0.y), pk2(h0.z, h0.w), pk2(h1.x, h1.y), pk2(h1.z, h1.w) };
    r.u1 = { pk2(h2.x, h2.y), pk2(h2.z, h2.w), pk2(h3.x, h3.y), pk2(h3.z, h3.w) };
    return r;
  };
  auto storeGU = [&](int buf, const Pre& r) {
    int o = buf * BTILE + ar * LDK + ac;
    *(v4u*)&sG[o]     = r.g0; *(v4u*)&sG[o + 8] = r.g1;
    *(v4u*)&sU[o]     = r.u0; *(v4u*)&sU[o + 8] = r.u1;
  };

  v8f accG[2][4], accU[2][4];
  v8f z = {};
#pragma unroll
  for (int i = 0; i < 2; ++i)
#pragma unroll
    for (int j = 0; j < 4; ++j) { accG[i][j] = z; accU[i][j] = z; }

  auto compute = [&](int cur) {
    const int cA = cur * ATILE;
    const int cB = cur * BTILE;
    FragBF aF[2];
#pragma unroll
    for (int i = 0; i < 2; ++i) {
      int row = cA + (wm + i * 16 + (lane & 15)) * LDK + ((lane >> 4) << 4);
      aF[i].u[0] = *(const v4u*)&sA[row];
      aF[i].u[1] = *(const v4u*)&sA[row + 8];
    }
#pragma unroll
    for (int j = 0; j < 4; ++j) {
      int row = cB + (wn + j * 16 + (lane & 15)) * LDK + ((lane >> 4) << 4);
      FragBF gF, uF;
      gF.u[0] = *(const v4u*)&sG[row];
      gF.u[1] = *(const v4u*)&sG[row + 8];
      uF.u[0] = *(const v4u*)&sU[row];
      uF.u[1] = *(const v4u*)&sU[row + 8];
#pragma unroll
      for (int i = 0; i < 2; ++i) {
        accG[i][j] = wmma_bf16(aF[i].v, gF.v, accG[i][j]);
        accU[i][j] = wmma_bf16(aF[i].v, uF.v, accU[i][j]);
      }
    }
  };

  // ---- pipeline prologue: fill buffer 0 ----
  issueA(0, 0);
  Pre p0 = loadGU(0);
  storeGU(0, p0);
  wait_async0();
  __syncthreads();

  const int steps = C / BK;
  for (int k = 0; k < steps - 1; ++k) {      // steady state: always prefetch next tile
    const int nxt = (k & 1) ^ 1;
    issueA(nxt, (k + 1) * BK);
    Pre pn = loadGU((k + 1) * BK);
    compute(k & 1);
    storeGU(nxt, pn);                        // buf nxt readers finished at the previous barrier
    wait_async0();
    __syncthreads();                         // single barrier per iteration
  }
  compute((steps - 1) & 1);                  // tail: compute only

  // SwiGLU epilogue -> a_buf bf16
#pragma unroll
  for (int i = 0; i < 2; ++i) {
#pragma unroll
    for (int j = 0; j < 4; ++j) {
      int n = n0 + wn + j * 16 + (lane & 15);
      float bg = b_in[(size_t)e * 2 * I + n];
      float bu = b_in[(size_t)e * 2 * I + I + n];
#pragma unroll
      for (int r = 0; r < 8; ++r) {
        int m = m0 + wm + i * 16 + ((lane >> 4) << 3) + r;
        if (m < Ne) {
          float g = accG[i][j][r] + bg;
          float u = accU[i][j][r] + bu;
          float a = g / (1.0f + __expf(-g)) * u;   // silu(g)*u
          a_buf[((size_t)e * T + m) * I + n] = f2bf(a);
        }
      }
    }
  }
}

// ---------------- GEMM2: y = a @ w_out^T, scale by gate, scatter-accumulate ----------------
__global__ __launch_bounds__(256) void moe_gemm2_kernel(const unsigned short* __restrict__ a_buf,
                                                        const float* __restrict__ w_out,
                                                        const float* __restrict__ b_out,
                                                        const int* __restrict__ counts,
                                                        const int* __restrict__ idxl,
                                                        const float* __restrict__ gvl,
                                                        float* __restrict__ out) {
  const int e  = blockIdx.z;
  const int m0 = blockIdx.x * BM;
  const int n0 = blockIdx.y * BN;   // over C
  const int Ne = counts[e];
  if (m0 >= Ne) return;

  __shared__ __align__(16) unsigned short sA[2 * ATILE];
  __shared__ __align__(16) unsigned short sB[2 * BTILE];

  const int tid  = threadIdx.x;
  const int lane = tid & 31;
  const int wv   = tid >> 5;
  const int wm   = (wv >> 1) << 5;
  const int wn   = (wv & 1) << 6;
  const int ar   = tid >> 1;
  const int ac   = (tid & 1) << 4;

  const bool av = (m0 + ar) < Ne;
  const int arow = av ? (m0 + ar) : m0;             // pad rows read a valid row; discarded later
  const unsigned short* aSrc = a_buf + ((size_t)e * T + arow) * I;
  const float* bRow = w_out + ((size_t)e * C + n0 + ar) * I;

  const unsigned aDst0 = lds_off(&sA[ar * LDK + ac]);

  auto issueA = [&](int buf, int k0) {
    unsigned d = aDst0 + (unsigned)(buf * ATILE * 2);
    const unsigned short* g = aSrc + k0 + ac;
    async_ld_b128(d,      g);
    async_ld_b128(d + 16, g + 8);
  };
  struct PreB { v4u b0, b1; };
  auto loadB = [&](int k0) {
    PreB r;
    const float4* s = (const float4*)(bRow + k0 + ac);
    float4 f0 = s[0], f1 = s[1], f2 = s[2], f3 = s[3];
    r.b0 = { pk2(f0.x, f0.y), pk2(f0.z, f0.w), pk2(f1.x, f1.y), pk2(f1.z, f1.w) };
    r.b1 = { pk2(f2.x, f2.y), pk2(f2.z, f2.w), pk2(f3.x, f3.y), pk2(f3.z, f3.w) };
    return r;
  };
  auto storeB = [&](int buf, const PreB& r) {
    int o = buf * BTILE + ar * LDK + ac;
    *(v4u*)&sB[o]     = r.b0;
    *(v4u*)&sB[o + 8] = r.b1;
  };

  v8f acc[2][4];
  v8f z = {};
#pragma unroll
  for (int i = 0; i < 2; ++i)
#pragma unroll
    for (int j = 0; j < 4; ++j) acc[i][j] = z;

  auto compute = [&](int cur) {
    const int cA = cur * ATILE;
    const int cB = cur * BTILE;
    FragBF aF[2];
#pragma unroll
    for (int i = 0; i < 2; ++i) {
      int row = cA + (wm + i * 16 + (lane & 15)) * LDK + ((lane >> 4) << 4);
      aF[i].u[0] = *(const v4u*)&sA[row];
      aF[i].u[1] = *(const v4u*)&sA[row + 8];
    }
#pragma unroll
    for (int j = 0; j < 4; ++j) {
      int row = cB + (wn + j * 16 + (lane & 15)) * LDK + ((lane >> 4) << 4);
      FragBF bF;
      bF.u[0] = *(const v4u*)&sB[row];
      bF.u[1] = *(const v4u*)&sB[row + 8];
#pragma unroll
      for (int i = 0; i < 2; ++i)
        acc[i][j] = wmma_bf16(aF[i].v, bF.v, acc[i][j]);
    }
  };

  issueA(0, 0);
  PreB q0 = loadB(0);
  storeB(0, q0);
  wait_async0();
  __syncthreads();

  const int steps = I / BK;
  for (int k = 0; k < steps - 1; ++k) {
    const int nxt = (k & 1) ^ 1;
    issueA(nxt, (k + 1) * BK);
    PreB qn = loadB((k + 1) * BK);
    compute(k & 1);
    storeB(nxt, qn);
    wait_async0();
    __syncthreads();
  }
  compute((steps - 1) & 1);

  // combine: out[t][c] += gate * (y + b_out[c])
#pragma unroll
  for (int i = 0; i < 2; ++i) {
#pragma unroll
    for (int j = 0; j < 4; ++j) {
      int n = n0 + wn + j * 16 + (lane & 15);
      float bo = b_out[(size_t)e * C + n];
#pragma unroll
      for (int r = 0; r < 8; ++r) {
        int m = m0 + wm + i * 16 + ((lane >> 4) << 3) + r;
        if (m < Ne) {
          int   t  = idxl[e * T + m];
          float gl = gvl[e * T + m];
          float val = (acc[i][j][r] + bo) * gl;
          __hip_atomic_fetch_add(&out[(size_t)t * C + n], val,
                                 __ATOMIC_RELAXED, __HIP_MEMORY_SCOPE_AGENT);
        }
      }
    }
  }
}

extern "C" void kernel_launch(void* const* d_in, const int* in_sizes, int n_in,
                              void* d_out, int out_size, void* d_ws, size_t ws_size,
                              hipStream_t stream) {
  const float* x     = (const float*)d_in[0];
  const float* rw    = (const float*)d_in[1];
  const float* w_in  = (const float*)d_in[2];
  const float* b_in  = (const float*)d_in[3];
  const float* w_out = (const float*)d_in[4];
  const float* b_out = (const float*)d_in[5];
  float* out = (float*)d_out;

  char* ws = (char*)d_ws;
  unsigned short* xb    = (unsigned short*)(ws);                       // 2 MB
  unsigned short* a_buf = (unsigned short*)(ws + (size_t)(2u << 20));  // 16 MB
  int*   counts = (int*)  (ws + (size_t)(18u << 20));                  // 32 B
  int*   idxl   = (int*)  (ws + (size_t)(18u << 20) + 256);            // 32 KB
  float* gvl    = (float*)(ws + (size_t)(18u << 20) + 256 + (size_t)E * T * 4);

  hipMemsetAsync(counts, 0, E * sizeof(int), stream);
  hipMemsetAsync(d_out, 0, (size_t)T * C * sizeof(float), stream);

  moe_cvt_kernel<<<(T * C / 4 + 255) / 256, 256, 0, stream>>>(x, xb);
  moe_router_kernel<<<T / 8, 256, 0, stream>>>(x, rw, counts, idxl, gvl);
  moe_gemm1_kernel<<<dim3(T / BM, I / BN, E), 256, 0, stream>>>(xb, w_in, b_in, counts, idxl, a_buf);
  moe_gemm2_kernel<<<dim3(T / BM, C / BN, E), 256, 0, stream>>>(a_buf, w_out, b_out, counts, idxl, gvl, out);
}
